// Attention_70901320123009
// MI455X (gfx1250) — matrix operand order repaired and checked by reading, not compiled, not run on hardware
//
#include <hip/hip_runtime.h>
#include <hip/hip_bf16.h>
#include <math.h>

typedef __attribute__((ext_vector_type(16))) __bf16 v16bf;
typedef __attribute__((ext_vector_type(8)))  __bf16 v8bf;
typedef __attribute__((ext_vector_type(4)))  __bf16 v4bf;
typedef __attribute__((ext_vector_type(8)))  float  v8f;

#define B_      64
#define T_      2048
#define D_      512
#define U_      512
#define TILE_M  128
#define LSTRIDE 520   // bf16 elems per LDS row (16B-aligned pad: kills bank conflicts)
#define TQ      8     // context T-split factor

__device__ __forceinline__ float fast_tanh(float x) {
#if __has_builtin(__builtin_amdgcn_tanhf)
    return __builtin_amdgcn_tanhf(x);      // v_tanh_f32 (gfx1250 TRANS op)
#else
    return tanhf(x);
#endif
}

// ---------------------------------------------------------------------------
// Kernel 0: build Wt in FRAGMENT-MAJOR bf16 layout.
//   chunk c = (n0/16)*16 + kc  (512 chunks of 1KB). Within chunk:
//   elem (lane l, e) = W1[k = kc*32 + (l>>4)*16 + e][n = (c>>4)*16 + (l&15)]
//   => B-fragment loads in the GEMM are contiguous 32B-per-lane streams.
// ---------------------------------------------------------------------------
__global__ __launch_bounds__(256) void prep_w1_kernel(const float* __restrict__ W1,
                                                      __bf16* __restrict__ Wt) {
    int o  = blockIdx.x * 256 + threadIdx.x;   // 262144 elems
    int e  = o & 15;
    int l  = (o >> 4) & 31;
    int c  = o >> 9;
    int kc = c & 15;
    int nc = c >> 4;
    int n  = nc * 16 + (l & 15);
    int k  = kc * 32 + (l >> 4) * 16 + e;
    Wt[o] = (__bf16)W1[(size_t)k * U_ + n];
}

// ---------------------------------------------------------------------------
// Kernel 1: comb[b][u] = hidden[b]@W2[:,u] + W2_b[u] + W1_b[u]
// ---------------------------------------------------------------------------
__global__ __launch_bounds__(256) void comb_kernel(const float* __restrict__ hidden,
                                                   const float* __restrict__ W2,
                                                   const float* __restrict__ W2b,
                                                   const float* __restrict__ W1b,
                                                   float* __restrict__ comb) {
    int b = blockIdx.x;
    int u = blockIdx.y * 256 + threadIdx.x;
    float acc = W2b[u] + W1b[u];
    const float* h = hidden + (size_t)b * D_;
    #pragma unroll 4
    for (int d = 0; d < D_; ++d)
        acc = fmaf(h[d], W2[(size_t)d * U_ + u], acc);   // coalesced over u
    comb[(size_t)b * U_ + u] = acc;
}

// ---------------------------------------------------------------------------
// Kernel 2: fused  logits[b,t] = sum_u Vw[u]*tanh( feat[b,t,:]@W1[:,u] + comb[b,u] )
//   grid (B, T/128), block 256 (8 waves x 16 T-rows), WMMA bf16 16x16x32
// ---------------------------------------------------------------------------
__global__ __launch_bounds__(256) void score_kernel(const float* __restrict__ features,
                                                    const __bf16* __restrict__ Wt,
                                                    const float* __restrict__ comb,
                                                    const float* __restrict__ Vw,
                                                    float* __restrict__ logits) {
    extern __shared__ __bf16 ldsA[];          // TILE_M * LSTRIDE bf16 = 130 KB
    const int b   = blockIdx.x;
    const int t0  = blockIdx.y * TILE_M;
    const int tid = threadIdx.x;

    // ---- stage feature tile f32 -> bf16 into LDS ----
    const float* fbase = features + ((size_t)b * T_ + t0) * D_;
    for (int idx = tid * 4; idx < TILE_M * D_; idx += 256 * 4) {
        int row = idx >> 9;
        int col = idx & 511;
        float4 v = *(const float4*)(fbase + (size_t)row * D_ + col);
        v4bf pk = { (__bf16)v.x, (__bf16)v.y, (__bf16)v.z, (__bf16)v.w };
        *(v4bf*)&ldsA[row * LSTRIDE + col] = pk;            // ds_store_b64
    }
    __syncthreads();

    const int wave    = tid >> 5;
    const int lane    = tid & 31;
    const int mrow    = lane & 15;      // A row / B column within tile
    const int hi      = lane >> 4;      // which K-half this lane holds
    const int rowbase = wave * 16;

    // ---- preload all 16 A fragments (K=512) into registers ----
    // A 16x32 bf16, contiguous K order used here (same as Wt): lanes 0-15 hold K {0..15}; lanes 16-31 {16..31}
    v16bf afrag[16];
    const int kloA = hi * 16;
    #pragma unroll
    for (int kc = 0; kc < 16; ++kc) {
        const __bf16* ap = &ldsA[(rowbase + mrow) * LSTRIDE + kc * 32 + kloA];
        v8bf lo  = *(const v8bf*)ap;          // K k0..k0+7
        v8bf hiv = *(const v8bf*)(ap + 8);    // K k0+8..k0+15
        afrag[kc] = __builtin_shufflevector(lo, hiv,
                     0,1,2,3,4,5,6,7,8,9,10,11,12,13,14,15);
    }

    // ---- N loop: 32 chunks of 16 columns; fused tanh + Vw reduction ----
    const __bf16* bbase = Wt + (size_t)lane * 16;   // fragment-major: coalesced
    float lpart[8];
    #pragma unroll
    for (int j = 0; j < 8; ++j) lpart[j] = 0.0f;

    for (int n0 = 0; n0 < U_; n0 += 16) {
        const int   n  = n0 + mrow;
        const float cb = comb[(size_t)b * U_ + n];
        const float vw = Vw[n];
        v8f acc = {};
        const __bf16* bp = bbase + (size_t)(n0 >> 4) * 16 * 512;
        #pragma unroll
        for (int kc = 0; kc < 16; ++kc) {
            v16bf bfrag = *(const v16bf*)(bp + (size_t)kc * 512);  // 32B/lane stream
            acc = __builtin_amdgcn_wmma_f32_16x16x32_bf16(
                      false, afrag[kc], false, bfrag, (short)0, acc, false, false);
        }
        #pragma unroll
        for (int j = 0; j < 8; ++j) {                // C vgpr j: row j + 8*hi, col n
            float s = fast_tanh(acc[j] + cb);
            lpart[j] = fmaf(s, vw, lpart[j]);
        }
    }

    // ---- reduce over the 16 columns held across each 16-lane half ----
    #pragma unroll
    for (int j = 0; j < 8; ++j) {
        float v = lpart[j];
        v += __shfl_xor(v, 1, 32);
        v += __shfl_xor(v, 2, 32);
        v += __shfl_xor(v, 4, 32);
        v += __shfl_xor(v, 8, 32);
        lpart[j] = v;
    }
    if (mrow == 0) {                                  // lanes 0 and 16 write
        const int rbase = t0 + rowbase + hi * 8;
        #pragma unroll
        for (int j = 0; j < 8; ++j)
            logits[(size_t)b * T_ + rbase + j] = lpart[j];
    }
}

// ---------------------------------------------------------------------------
// Kernel 3: softmax over T per batch  (V_b dropped: shift-invariant)
// ---------------------------------------------------------------------------
__global__ __launch_bounds__(256) void softmax_kernel(const float* __restrict__ logits,
                                                      float* __restrict__ attn) {
    __shared__ float red[8];
    const int b = blockIdx.x, tid = threadIdx.x;
    const float* lp = logits + (size_t)b * T_;
    float v[8], m = -3.0e38f;
    #pragma unroll
    for (int j = 0; j < 8; ++j) { v[j] = lp[tid + j * 256]; m = fmaxf(m, v[j]); }
    #pragma unroll
    for (int off = 16; off; off >>= 1) m = fmaxf(m, __shfl_xor(m, off, 32));
    if ((tid & 31) == 0) red[tid >> 5] = m;
    __syncthreads();
    m = red[0];
    #pragma unroll
    for (int w = 1; w < 8; ++w) m = fmaxf(m, red[w]);
    float s = 0.0f;
    #pragma unroll
    for (int j = 0; j < 8; ++j) { v[j] = __expf(v[j] - m); s += v[j]; }
    #pragma unroll
    for (int off = 16; off; off >>= 1) s += __shfl_xor(s, off, 32);
    __syncthreads();
    if ((tid & 31) == 0) red[tid >> 5] = s;
    __syncthreads();
    s = 0.0f;
    #pragma unroll
    for (int w = 0; w < 8; ++w) s += red[w];
    const float inv = 1.0f / s;
    float* ap = attn + (size_t)b * T_;
    #pragma unroll
    for (int j = 0; j < 8; ++j) ap[tid + j * 256] = v[j] * inv;
}

// ---------------------------------------------------------------------------
// Kernel 4a: partial context over T-slices (512 blocks for HBM saturation)
// ---------------------------------------------------------------------------
__global__ __launch_bounds__(256) void context_part_kernel(const float* __restrict__ features,
                                                           const float* __restrict__ attn,
                                                           float* __restrict__ part) {
    const int b = blockIdx.x, q = blockIdx.y, tid = threadIdx.x;  // 256 thr x float2
    const float2* f = (const float2*)(features + ((size_t)b * T_ + (size_t)q * (T_ / TQ)) * D_);
    const float*  ap = attn + (size_t)b * T_ + (size_t)q * (T_ / TQ);
    float2 acc = make_float2(0.0f, 0.0f);
    for (int t = 0; t < T_ / TQ; ++t) {
        float  a = ap[t];
        float2 v = f[(size_t)t * 256 + tid];
        acc.x = fmaf(a, v.x, acc.x);
        acc.y = fmaf(a, v.y, acc.y);
    }
    ((float2*)part)[((size_t)b * TQ + q) * 256 + tid] = acc;
}

// Kernel 4b: deterministic reduction of TQ partials -> context
__global__ __launch_bounds__(256) void context_reduce_kernel(const float* __restrict__ part,
                                                             float* __restrict__ ctx) {
    int i = blockIdx.x * 256 + threadIdx.x;      // 32768 elems
    int b = i >> 9, d = i & 511;
    float s = 0.0f;
    #pragma unroll
    for (int q = 0; q < TQ; ++q) s += part[((size_t)b * TQ + q) * D_ + d];
    ctx[i] = s;
}

// ---------------------------------------------------------------------------
extern "C" void kernel_launch(void* const* d_in, const int* in_sizes, int n_in,
                              void* d_out, int out_size, void* d_ws, size_t ws_size,
                              hipStream_t stream) {
    const float* features = (const float*)d_in[0];   // [B,T,D]
    const float* hidden   = (const float*)d_in[1];   // [B,D]
    const float* W1_w     = (const float*)d_in[2];   // [D,U]
    const float* W1_b     = (const float*)d_in[3];   // [U]
    const float* W2_w     = (const float*)d_in[4];   // [D,U]
    const float* W2_b     = (const float*)d_in[5];   // [U]
    const float* V_w      = (const float*)d_in[6];   // [U,1]
    // d_in[7] = V_b : softmax shift-invariant, unused.

    float* ctx  = (float*)d_out;                     // [B,D]   32768
    float* attn = (float*)d_out + B_ * D_;           // [B,T,1] 131072

    // workspace: Wt bf16 (512KB) | comb (128KB) | logits (512KB) | part (1MB)
    __bf16* Wt     = (__bf16*)d_ws;
    float*  comb   = (float*)((char*)d_ws + (size_t)D_ * U_ * sizeof(__bf16));
    float*  logits = comb + (size_t)B_ * U_;
    float*  part   = logits + (size_t)B_ * T_;

    prep_w1_kernel<<<dim3((D_ * U_) / 256), 256, 0, stream>>>(W1_w, Wt);
    comb_kernel<<<dim3(B_, U_ / 256), 256, 0, stream>>>(hidden, W2_w, W2_b, W1_b, comb);

    const size_t lds_bytes = (size_t)TILE_M * LSTRIDE * sizeof(__bf16);  // 133120
    score_kernel<<<dim3(B_, T_ / TILE_M), 256, lds_bytes, stream>>>(
        features, Wt, comb, V_w, logits);

    softmax_kernel<<<dim3(B_), 256, 0, stream>>>(logits, attn);
    context_part_kernel<<<dim3(B_, TQ), 256, 0, stream>>>(features, attn, part);
    context_reduce_kernel<<<dim3((B_ * D_) / 256), 256, 0, stream>>>(part, ctx);
}
